// MultiHeadSelfAttention_67723044323583
// MI455X (gfx1250) — compile-verified, matrix operand order checked by reading, not run on hardware
//
#include <hip/hip_runtime.h>
#include <hip/hip_bf16.h>

typedef __bf16 bf16_t;
typedef __attribute__((ext_vector_type(16))) __bf16 v16bf;
typedef __attribute__((ext_vector_type(8)))  __bf16 v8bf;
typedef __attribute__((ext_vector_type(8)))  float  v8f;
typedef __attribute__((ext_vector_type(4)))  float  f32x4;
typedef __attribute__((ext_vector_type(4)))  unsigned int u32x4;
typedef int i32x4 __attribute__((__vector_size__(4 * sizeof(int))));

#define TVAL 2048
#define CVAL 1024
#define HVAL 16
#define DHVAL 64

// LDS row strides (elements) chosen 16B-aligned and bank-conflict-friendly
#define KSTR 72   // K tile rows: 64 d elems + pad
#define VSTR 40   // V tile rows: 32 t elems + pad
#define BSTR 40   // GEMM B tile rows: 32 k elems + pad
#define PSTR 48   // P tile rows: 32 cols + pad

#if defined(__HIP_DEVICE_COMPILE__) && \
    __has_builtin(__builtin_amdgcn_global_load_async_to_lds_b128) && \
    __has_builtin(__builtin_amdgcn_s_wait_asynccnt)
#define HAVE_ASYNC 1
#else
#define HAVE_ASYNC 0
#endif

__device__ __forceinline__ void async_copy16(const bf16_t* g, bf16_t* l) {
#if HAVE_ASYNC
  __builtin_amdgcn_global_load_async_to_lds_b128(
      (i32x4*)(void*)g, (i32x4*)(void*)l, 0, 0);
#else
  *(u32x4*)l = *(const u32x4*)g;
#endif
}

template <int N>
__device__ __forceinline__ void async_wait() {
#if HAVE_ASYNC
  __builtin_amdgcn_s_wait_asynccnt(N);
#endif
}

union BF16x16 { v16bf v; v8bf h[2]; };

__device__ __forceinline__ v8f wmma_bf16(v16bf a, v16bf b, v8f c) {
  return __builtin_amdgcn_wmma_f32_16x16x32_bf16(false, a, false, b, (short)0, c, false, false);
}

// A/B fragment from bf16 row-major: lane holds row/col (lane&15);
// element i -> k = base + half*8 + (i<8 ? i : 8+i). k0/k1 = the two chunk offsets.
__device__ __forceinline__ v16bf load_b16x16(const bf16_t* row, int k0, int k1) {
  BF16x16 u;
  u.h[0] = *(const v8bf*)(row + k0);
  u.h[1] = *(const v8bf*)(row + k1);
  return u.v;
}

// A-fragment from fp32 source (convert to bf16 in registers)
__device__ __forceinline__ v16bf load_a_from_f32(const float* row, int k0, int k1) {
  f32x4 x0 = *(const f32x4*)(row + k0);
  f32x4 x1 = *(const f32x4*)(row + k0 + 4);
  f32x4 x2 = *(const f32x4*)(row + k1);
  f32x4 x3 = *(const f32x4*)(row + k1 + 4);
  v16bf a;
#pragma unroll
  for (int i = 0; i < 4; i++) {
    a[i]      = (bf16_t)x0[i];
    a[4 + i]  = (bf16_t)x1[i];
    a[8 + i]  = (bf16_t)x2[i];
    a[12 + i] = (bf16_t)x3[i];
  }
  return a;
}

// ---------------- Kernel 0: weight conversion f32 -> bf16 ----------------
__global__ void __launch_bounds__(256) cvt_weights(
    const float* __restrict__ wqkv, const float* __restrict__ wo,
    bf16_t* __restrict__ wqkv_bf, bf16_t* __restrict__ wo_bf) {
  const int NQ = 3 * CVAL * CVAL;
  const int NO = CVAL * CVAL;
  int stride = gridDim.x * 256;
  for (int i = blockIdx.x * 256 + threadIdx.x; i < NQ; i += stride)
    wqkv_bf[i] = (bf16_t)wqkv[i];
  for (int i = blockIdx.x * 256 + threadIdx.x; i < NO; i += stride)
    wo_bf[i] = (bf16_t)wo[i];
}

// ---------------- Kernel 1: QKV projection --------------------------------
// out[m][n] = sum_k x[m][k]*Wqkv[n][k] + bqkv[n]; scatter into Q,K,Vt (bf16).
// Block = 8 waves sharing one 64-col weight slab (staged via async->LDS,
// double-buffered); each wave owns a 32(M)x64(N) tile -> 8 WMMA per K-step.
__global__ void __launch_bounds__(256) qkv_kernel(
    const float* __restrict__ x,       // [8192][1024] f32
    const bf16_t* __restrict__ wq,     // [3072][1024] bf16
    const float* __restrict__ bqkv,    // [3072]
    bf16_t* __restrict__ Qb,           // [B,H,T,Dh], pre-scaled by 1/8
    bf16_t* __restrict__ Kb,           // [B,H,T,Dh]
    bf16_t* __restrict__ Vt) {         // [B,H,Dh,T]
  __shared__ __align__(16) bf16_t btile[2][64 * BSTR];
  const int lane = threadIdx.x & 31, wid = threadIdx.x >> 5;
  const int half = lane >> 4, l16 = lane & 15;
  const int tid = threadIdx.x;
  int wg = blockIdx.x * 8 + wid;
  int mt = wg & 255;     // 256 M tiles of 32 rows
  int nc = wg >> 8;      // 48 N chunks of 64 (uniform across the block)
  int mbase = mt * 32, nbase = nc * 64;
  const float* arow0 = x + (size_t)(mbase + l16) * CVAL;
  const float* arow1 = x + (size_t)(mbase + 16 + l16) * CVAL;
  const int brow_ld = tid >> 2, bcol_ld = (tid & 3) * 8;   // 64 rows x 4 chunks

  v8f acc0[4] = {}, acc1[4] = {};
  int cur = 0;
  async_copy16(wq + (size_t)(nbase + brow_ld) * CVAL + bcol_ld,
               &btile[0][brow_ld * BSTR + bcol_ld]);
  for (int kb = 0; kb < CVAL; kb += 32) {
    bool has_next = (kb + 32) < CVAL;
    if (has_next)
      async_copy16(wq + (size_t)(nbase + brow_ld) * CVAL + (kb + 32) + bcol_ld,
                   &btile[cur ^ 1][brow_ld * BSTR + bcol_ld]);
    if (has_next) async_wait<1>(); else async_wait<0>();
    __syncthreads();
    int k0 = kb + half * 8, k1 = kb + 16 + half * 8;
    v16bf a0 = load_a_from_f32(arow0, k0, k1);
    v16bf a1 = load_a_from_f32(arow1, k0, k1);
    int lk0 = half * 8, lk1 = 16 + half * 8;
#pragma unroll
    for (int nt = 0; nt < 4; nt++) {
      const bf16_t* brow = &btile[cur][(nt * 16 + l16) * BSTR];
      v16bf bf = load_b16x16(brow, lk0, lk1);
      acc0[nt] = wmma_bf16(a0, bf, acc0[nt]);
      acc1[nt] = wmma_bf16(a1, bf, acc1[nt]);
    }
    __syncthreads();
    cur ^= 1;
  }

  // scatter epilogue: s, hh are wave-uniform closed forms of nc
  int s = nc >> 4;           // 0=q 1=k 2=v
  int hh = nc & 15;
  int bidx = mbase >> 11;
  int tb = mbase & 2047;
  size_t hbase = (size_t)(bidx * HVAL + hh);
#pragma unroll
  for (int nt = 0; nt < 4; nt++) {
    int d = nt * 16 + l16;
    float bias = bqkv[nc * 64 + d];
#pragma unroll
    for (int r = 0; r < 8; r++) {
      int t0 = tb + r + half * 8, t1 = t0 + 16;
      float v0 = acc0[nt][r] + bias;
      float v1 = acc1[nt][r] + bias;
      if (s == 0) {
        Qb[(hbase * TVAL + t0) * DHVAL + d] = (bf16_t)(v0 * 0.125f);
        Qb[(hbase * TVAL + t1) * DHVAL + d] = (bf16_t)(v1 * 0.125f);
      } else if (s == 1) {
        Kb[(hbase * TVAL + t0) * DHVAL + d] = (bf16_t)v0;
        Kb[(hbase * TVAL + t1) * DHVAL + d] = (bf16_t)v1;
      } else {
        Vt[(hbase * DHVAL + d) * TVAL + t0] = (bf16_t)v0;
        Vt[(hbase * DHVAL + d) * TVAL + t1] = (bf16_t)v1;
      }
    }
  }
}

// ---------------- Kernel 2: causal flash attention ------------------------
// Block = 8 waves, same (b,h), consecutive 16-query tiles. K/V 32-key tiles
// staged cooperatively via async->LDS (double-buffered); waves compute only
// their causal prefix; all waves run the shared load/barrier schedule.
__global__ void __launch_bounds__(256) attn_kernel(
    const bf16_t* __restrict__ Qb, const bf16_t* __restrict__ Kb,
    const bf16_t* __restrict__ Vt, bf16_t* __restrict__ ctx) {
  __shared__ __align__(16) bf16_t ktile[2][32 * KSTR];
  __shared__ __align__(16) bf16_t vtile[2][64 * VSTR];
  __shared__ __align__(16) bf16_t plds[8][16 * PSTR];
  const int lane = threadIdx.x & 31, wid = threadIdx.x >> 5;
  const int half = lane >> 4, l16 = lane & 15;
  const int tid = threadIdx.x;
  int wg = blockIdx.x * 8 + wid;
  int qt = wg & 127;
  int h = (wg >> 7) & 15;
  int b = wg >> 11;
  int qbase = qt * 16;
  int qt_base = (blockIdx.x * 8) & 127;                  // first qt in this block
  int my_nblk = ((qbase + 15) >> 5) + 1;
  int nblk_max = (((qt_base + 7) * 16 + 15) >> 5) + 1;   // last wave's count

  const bf16_t* qrow = Qb + ((size_t)(b * HVAL + h) * TVAL + qbase + l16) * DHVAL;
  v16bf qf0 = load_b16x16(qrow, half * 8, 16 + half * 8);        // d 0..31
  v16bf qf1 = load_b16x16(qrow, 32 + half * 8, 48 + half * 8);   // d 32..63

  v8f acc[4] = {};
  float rmax[8], rsum[8];
#pragma unroll
  for (int r = 0; r < 8; r++) { rmax[r] = -3.0e38f; rsum[r] = 0.f; }

  bf16_t* pw = &plds[wid][0];
  const bf16_t* kbptr = Kb + (size_t)(b * HVAL + h) * TVAL * DHVAL;
  const bf16_t* vbptr = Vt + (size_t)(b * HVAL + h) * DHVAL * TVAL;

  // cooperative K/V tile loaders: one 16B piece of K and of V per thread
  const int kr = tid >> 3, kc = (tid & 7) * 8;   // K: 32 rows x 8 chunks
  const int vr = tid >> 2, vc = (tid & 3) * 8;   // V: 64 rows x 4 chunks

  int cur = 0;
  async_copy16(kbptr + (size_t)kr * DHVAL + kc, &ktile[0][kr * KSTR + kc]);
  async_copy16(vbptr + (size_t)vr * TVAL + vc, &vtile[0][vr * VSTR + vc]);

  for (int jb = 0; jb < nblk_max; jb++) {
    int kb0 = jb * 32;
    bool has_next = (jb + 1) < nblk_max;
    if (has_next) {
      int kn = kb0 + 32;
      async_copy16(kbptr + (size_t)(kn + kr) * DHVAL + kc,
                   &ktile[cur ^ 1][kr * KSTR + kc]);
      async_copy16(vbptr + (size_t)vr * TVAL + kn + vc,
                   &vtile[cur ^ 1][vr * VSTR + vc]);
    }
    if (has_next) async_wait<2>(); else async_wait<0>();
    __syncthreads();

    if (jb < my_nblk) {
      const bf16_t* kl = &ktile[cur][0];
      const bf16_t* vl = &vtile[cur][0];
      // ---- S = Q * K^T for 16q x 32keys (4 WMMA) ----
      v8f s[2];
#pragma unroll
      for (int kt = 0; kt < 2; kt++) {
        const bf16_t* krow = kl + (kt * 16 + l16) * KSTR;
        v16bf b0 = load_b16x16(krow, half * 8, 16 + half * 8);
        v16bf b1 = load_b16x16(krow, 32 + half * 8, 48 + half * 8);
        v8f z = {};
        z = wmma_bf16(qf0, b0, z);
        z = wmma_bf16(qf1, b1, z);
        s[kt] = z;
      }
      // ---- causal mask + online softmax (row = r + half*8, col = l16) ----
#pragma unroll
      for (int r = 0; r < 8; r++) {
        int qg = qbase + r + half * 8;
        float s0 = (kb0 + l16 <= qg) ? s[0][r] : -3.0e38f;
        float s1 = (kb0 + 16 + l16 <= qg) ? s[1][r] : -3.0e38f;
        float m = fmaxf(s0, s1);
        m = fmaxf(m, __shfl_xor(m, 1));
        m = fmaxf(m, __shfl_xor(m, 2));
        m = fmaxf(m, __shfl_xor(m, 4));
        m = fmaxf(m, __shfl_xor(m, 8));   // reduces within each 16-lane half
        float nm = fmaxf(rmax[r], m);
        float f = __expf(rmax[r] - nm);
        rmax[r] = nm;
        float p0 = __expf(s0 - nm);
        float p1 = __expf(s1 - nm);
        float ps = p0 + p1;
        ps += __shfl_xor(ps, 1);
        ps += __shfl_xor(ps, 2);
        ps += __shfl_xor(ps, 4);
        ps += __shfl_xor(ps, 8);
        rsum[r] = rsum[r] * f + ps;
#pragma unroll
        for (int nt = 0; nt < 4; nt++) acc[nt][r] *= f;
        int row = r + half * 8;
        pw[row * PSTR + l16] = (bf16_t)p0;
        pw[row * PSTR + 16 + l16] = (bf16_t)p1;
      }
      // ---- reload P as A-fragment (DS ops in-order within a wave) ----
      const bf16_t* prow = pw + l16 * PSTR;
      BF16x16 pu;
      pu.h[0] = *(const v8bf*)(prow + half * 8);
      pu.h[1] = *(const v8bf*)(prow + 16 + half * 8);
      v16bf pf = pu.v;
      // ---- acc += P * V (4 WMMA) ----
#pragma unroll
      for (int nt = 0; nt < 4; nt++) {
        const bf16_t* vrow = vl + (nt * 16 + l16) * VSTR;
        v16bf bv = load_b16x16(vrow, half * 8, 16 + half * 8);
        acc[nt] = wmma_bf16(pf, bv, acc[nt]);
      }
    }
    __syncthreads();
    cur ^= 1;
  }

  // ---- normalize and write ctx[b][t][h*64 + d] ----
#pragma unroll
  for (int r = 0; r < 8; r++) {
    float inv = 1.0f / rsum[r];
    int t = qbase + r + half * 8;
#pragma unroll
    for (int nt = 0; nt < 4; nt++) {
      ctx[(size_t)(b * TVAL + t) * CVAL + h * DHVAL + nt * 16 + l16] =
          (bf16_t)(acc[nt][r] * inv);
    }
  }
}

// ---------------- Kernel 3: output projection -----------------------------
// out[m][n] = sum_k ctx[m][k]*Wo[n][k] + bo[n] (fp32 out). Same staging as QKV.
__global__ void __launch_bounds__(256) proj_kernel(
    const bf16_t* __restrict__ ctx, const bf16_t* __restrict__ wo,
    const float* __restrict__ bo, float* __restrict__ out) {
  __shared__ __align__(16) bf16_t btile[2][64 * BSTR];
  const int lane = threadIdx.x & 31, wid = threadIdx.x >> 5;
  const int half = lane >> 4, l16 = lane & 15;
  const int tid = threadIdx.x;
  int wg = blockIdx.x * 8 + wid;
  int mt = wg & 255;     // 256 M tiles of 32 rows
  int nc = wg >> 8;      // 16 N chunks of 64 (uniform per block)
  int mbase = mt * 32, nbase = nc * 64;
  const bf16_t* arow0 = ctx + (size_t)(mbase + l16) * CVAL;
  const bf16_t* arow1 = ctx + (size_t)(mbase + 16 + l16) * CVAL;
  const int brow_ld = tid >> 2, bcol_ld = (tid & 3) * 8;

  v8f acc0[4] = {}, acc1[4] = {};
  int cur = 0;
  async_copy16(wo + (size_t)(nbase + brow_ld) * CVAL + bcol_ld,
               &btile[0][brow_ld * BSTR + bcol_ld]);
  for (int kb = 0; kb < CVAL; kb += 32) {
    bool has_next = (kb + 32) < CVAL;
    if (has_next)
      async_copy16(wo + (size_t)(nbase + brow_ld) * CVAL + (kb + 32) + bcol_ld,
                   &btile[cur ^ 1][brow_ld * BSTR + bcol_ld]);
    if (has_next) async_wait<1>(); else async_wait<0>();
    __syncthreads();
    int k0 = kb + half * 8, k1 = kb + 16 + half * 8;
    v16bf a0 = load_b16x16(arow0, k0, k1);
    v16bf a1 = load_b16x16(arow1, k0, k1);
    int lk0 = half * 8, lk1 = 16 + half * 8;
#pragma unroll
    for (int nt = 0; nt < 4; nt++) {
      const bf16_t* brow = &btile[cur][(nt * 16 + l16) * BSTR];
      v16bf bf = load_b16x16(brow, lk0, lk1);
      acc0[nt] = wmma_bf16(a0, bf, acc0[nt]);
      acc1[nt] = wmma_bf16(a1, bf, acc1[nt]);
    }
    __syncthreads();
    cur ^= 1;
  }
#pragma unroll
  for (int nt = 0; nt < 4; nt++) {
    int n = nbase + nt * 16 + l16;
    float bias = bo[n];
#pragma unroll
    for (int r = 0; r < 8; r++) {
      int m0 = mbase + r + half * 8;
      out[(size_t)m0 * CVAL + n] = acc0[nt][r] + bias;
      out[(size_t)(m0 + 16) * CVAL + n] = acc1[nt][r] + bias;
    }
  }
}

extern "C" void kernel_launch(void* const* d_in, const int* in_sizes, int n_in,
                              void* d_out, int out_size, void* d_ws, size_t ws_size,
                              hipStream_t stream) {
  (void)in_sizes; (void)n_in; (void)out_size; (void)ws_size;
  const float* x    = (const float*)d_in[0];
  const float* Wqkv = (const float*)d_in[1];
  const float* bqkv = (const float*)d_in[2];
  const float* Wo   = (const float*)d_in[3];
  const float* bo   = (const float*)d_in[4];
  float* out = (float*)d_out;

  char* ws = (char*)d_ws;
  bf16_t* wqkv_bf = (bf16_t*)(ws);                 //  6 MB: [3072][1024]
  bf16_t* wo_bf   = (bf16_t*)(ws +  6291456);      //  2 MB: [1024][1024]
  bf16_t* Qb      = (bf16_t*)(ws +  8388608);      // 16 MB: [B,H,T,Dh]
  bf16_t* Kb      = (bf16_t*)(ws + 25165824);      // 16 MB: [B,H,T,Dh]
  bf16_t* Vt      = (bf16_t*)(ws + 41943040);      // 16 MB: [B,H,Dh,T]
  bf16_t* ctx     = (bf16_t*)(ws + 58720256);      // 16 MB: [B,T,C]

  cvt_weights<<<2048, 256, 0, stream>>>(Wqkv, Wo, wqkv_bf, wo_bf);
  // 256 M-tiles * 48 N-chunks = 12288 waves / 8 per block
  qkv_kernel<<<1536, 256, 0, stream>>>(x, wqkv_bf, bqkv, Qb, Kb, Vt);
  // 4*16*128 = 8192 waves / 8 per block
  attn_kernel<<<1024, 256, 0, stream>>>(Qb, Kb, Vt, ctx);
  // 256 M-tiles * 16 N-chunks = 4096 waves / 8 per block
  proj_kernel<<<512, 256, 0, stream>>>(ctx, wo_bf, bo, out);
}
